// ECForGraphTCN_5171140624472
// MI455X (gfx1250) — matrix operand order repaired and checked by reading, not compiled, not run on hardware
//
#include <hip/hip_runtime.h>
#include <hip/hip_bf16.h>

typedef _Float16 half_t;
typedef __attribute__((ext_vector_type(16))) _Float16 v16h;
typedef __attribute__((ext_vector_type(8)))  float    v8f;

static constexpr int NNODES  = 50000;
static constexpr int NEDGES  = 800000;
static constexpr int NODE_IN = 14;
static constexpr int EDGE_IN = 4;

// ---------------------------------------------------------------------------
// Register-blocked WMMA GEMM: one wave computes MT*16 rows x N cols.
// For each (n0,k0): load B fragment ONCE, apply to MT A-tiles (4x L2 reuse).
// A frag (16-bit A 16x32, ISA 7.12.2): lane&15 = M row; lane>=16 -> K 8..15/24..31;
//   v16h elems 0..7 = K k0..k0+7, elems 8..15 = K k0+16..k0+23.
// B frag: lane&15 = N column n0+(lane&15); elems hold K rows in A's k order (stride ldw).
// C/D: VGPR r, lane l -> (M = r + 8*(l>=16), N = l&15).
// ---------------------------------------------------------------------------
template<int MT>
__device__ __forceinline__ void gemm_acc_mt(const half_t* __restrict__ X, int ldx,
                                            const half_t* __restrict__ W, int ldw,
                                            float binit, int K, int n0, v8f* acc)
{
    const int lane = threadIdx.x & 31;
    const int hi   = lane >> 4;
    const int nl   = lane & 15;
#pragma unroll
    for (int t = 0; t < MT; ++t)
#pragma unroll
        for (int r = 0; r < 8; ++r) acc[t][r] = binit;
    for (int k0 = 0; k0 < K; k0 += 32) {
        v16h b;
        const half_t* pb = W + (size_t)(k0 + hi * 8) * ldw + n0 + nl;
#pragma unroll
        for (int j = 0; j < 8; ++j) { b[j] = pb[j * ldw]; b[8 + j] = pb[(16 + j) * ldw]; }
#pragma unroll
        for (int t = 0; t < MT; ++t) {
            v16h a;
            const half_t* pa = X + (t * 16 + nl) * ldx + k0 + hi * 8;
#pragma unroll
            for (int i = 0; i < 8; ++i) { a[i] = pa[i]; a[8 + i] = pa[16 + i]; }
            acc[t] = __builtin_amdgcn_wmma_f32_16x16x32_f16(false, a, false, b,
                                                            (short)0, acc[t], false, false);
        }
    }
}

// Y[MT*16, N] = relu(X @ W + b); stores guarded by row < mvalid.
template<int MT>
__device__ __forceinline__ void gemm_relu_mt(const half_t* __restrict__ X, int ldx,
                                             const half_t* __restrict__ W,
                                             const float* __restrict__ bias,
                                             half_t* __restrict__ Y, int ldy,
                                             int K, int N, int mvalid)
{
    const int lane = threadIdx.x & 31;
    const int hi   = lane >> 4;
    const int nl   = lane & 15;
    for (int n0 = 0; n0 < N; n0 += 16) {
        float bv = bias ? bias[n0 + nl] : 0.0f;
        v8f acc[MT];
        gemm_acc_mt<MT>(X, ldx, W, N, bv, K, n0, acc);
#pragma unroll
        for (int t = 0; t < MT; ++t)
#pragma unroll
            for (int r = 0; r < 8; ++r) {
                int row = t * 16 + r + hi * 8;
                if (row < mvalid)
                    Y[row * ldy + n0 + nl] = (half_t)fmaxf(acc[t][r], 0.0f);
            }
    }
}

// ---------------------------------------------------------------------------
// Encoders: [64,fin] -> pad K=32 -> 128 (relu) -> 64 (relu), no bias.
// ---------------------------------------------------------------------------
__global__ __launch_bounds__(32) void k_encoder(const float* __restrict__ xin, int fin,
                                                int nrows,
                                                const half_t* __restrict__ W0,
                                                const half_t* __restrict__ W1,
                                                half_t* __restrict__ out)
{
    __shared__ __align__(16) half_t sIn[64 * 32];
    __shared__ __align__(16) half_t sHid[64 * 128];
    const int lane = threadIdx.x;
    const int base = blockIdx.x * 64;
    for (int idx = lane; idx < 64 * 32; idx += 32) sIn[idx] = (half_t)0.0f;
    __syncthreads();
    for (int idx = lane; idx < 64 * fin; idx += 32) {
        int m = idx / fin, f = idx - m * fin;
        int row = min(base + m, nrows - 1);
        sIn[m * 32 + f] = (half_t)xin[(size_t)row * fin + f];
    }
    __syncthreads();
    gemm_relu_mt<4>(sIn, 32, W0, nullptr, sHid, 128, 32, 128, 64);
    __syncthreads();
    int mvalid = min(64, nrows - base);
    gemm_relu_mt<4>(sHid, 128, W1, nullptr, out + (size_t)base * 64, 64, 128, 64, mvalid);
}

// ---------------------------------------------------------------------------
// Edge update (rel MLP 192->128->128->64) + atomic segment_sum + e residual.
// m_in = [h[dst], h[src], e] ; 64 edges per wave. sB aliases sIn.
// ---------------------------------------------------------------------------
__global__ __launch_bounds__(32) void k_edge_layer(
    const int* __restrict__ ei, const half_t* __restrict__ h16,
    const half_t* __restrict__ e_in,
    const half_t* __restrict__ W0, const float* __restrict__ b0,
    const half_t* __restrict__ W1, const float* __restrict__ b1,
    const half_t* __restrict__ W2, const float* __restrict__ b2,
    float* __restrict__ agg, half_t* __restrict__ e_out)
{
    __shared__ int sSrc[64], sDst[64];
    __shared__ __align__(16) half_t sIn[64 * 192];   // also reused as sB
    __shared__ __align__(16) half_t sA[64 * 128];
    half_t* sB = sIn;
    const int lane = threadIdx.x;
    const int base = blockIdx.x * 64;
    for (int i = lane; i < 64; i += 32) {
        sSrc[i] = ei[base + i];
        sDst[i] = ei[NEDGES + base + i];
    }
    __syncthreads();
    // gather concat input: 24 16-byte chunks per row
    for (int idx = lane; idx < 64 * 24; idx += 32) {
        int m = idx / 24, c = idx - m * 24;
        uint4 v;
        if (c < 8)       v = ((const uint4*)(h16 + (size_t)sDst[m] * 64))[c];
        else if (c < 16) v = ((const uint4*)(h16 + (size_t)sSrc[m] * 64))[c - 8];
        else             v = ((const uint4*)(e_in + (size_t)(base + m) * 64))[c - 16];
        ((uint4*)(sIn + m * 192))[c] = v;
    }
    __syncthreads();
    gemm_relu_mt<4>(sIn, 192, W0, b0, sA, 128, 192, 128, 64);
    __syncthreads();
    gemm_relu_mt<4>(sA, 128, W1, b1, sB, 128, 128, 128, 64);
    __syncthreads();
    const int hi = lane >> 4, nl = lane & 15;
    for (int n0 = 0; n0 < 64; n0 += 16) {
        v8f acc[4];
        gemm_acc_mt<4>(sB, 128, W2, 64, b2[n0 + nl], 128, n0, acc);
#pragma unroll
        for (int t = 0; t < 4; ++t)
#pragma unroll
            for (int r = 0; r < 8; ++r) {
                int   m  = t * 16 + r + hi * 8;
                int   n  = n0 + nl;
                float de = acc[t][r];                       // raw delta_e aggregates
                atomicAdd(&agg[(size_t)sDst[m] * 64 + n], de);
                float eo = (float)e_in[(size_t)(base + m) * 64 + n];
                e_out[(size_t)(base + m) * 64 + n] =
                    (half_t)(0.5f * eo + 0.5f * fmaxf(de, 0.0f));
            }
    }
}

// ---------------------------------------------------------------------------
// Node update (obj MLP 128->128->128->64) + h residual, in-place on h16.
// in = [h, agg] ; 64 nodes per wave (tail-guarded). sB aliases sIn.
// ---------------------------------------------------------------------------
__global__ __launch_bounds__(32) void k_node_layer(
    half_t* __restrict__ h16, const float* __restrict__ agg,
    const half_t* __restrict__ W0, const float* __restrict__ b0,
    const half_t* __restrict__ W1, const float* __restrict__ b1,
    const half_t* __restrict__ W2, const float* __restrict__ b2)
{
    __shared__ __align__(16) half_t sIn[64 * 128];   // also reused as sB
    __shared__ __align__(16) half_t sA[64 * 128];
    half_t* sB = sIn;
    const int lane = threadIdx.x;
    const int base = blockIdx.x * 64;
    // h part: 8 16-byte chunks per row
    for (int idx = lane; idx < 64 * 8; idx += 32) {
        int m = idx >> 3, c = idx & 7;
        int row = min(base + m, NNODES - 1);
        ((uint4*)(sIn + m * 128))[c] = ((const uint4*)(h16 + (size_t)row * 64))[c];
    }
    // agg part: 16 chunks of 4 floats per row, converted to f16
    for (int idx = lane; idx < 64 * 16; idx += 32) {
        int m = idx >> 4, c = idx & 15;
        int row = min(base + m, NNODES - 1);
        float4 v = ((const float4*)(agg + (size_t)row * 64))[c];
        half_t* p = sIn + m * 128 + 64 + c * 4;
        p[0] = (half_t)v.x; p[1] = (half_t)v.y; p[2] = (half_t)v.z; p[3] = (half_t)v.w;
    }
    __syncthreads();
    gemm_relu_mt<4>(sIn, 128, W0, b0, sA, 128, 128, 128, 64);
    __syncthreads();
    gemm_relu_mt<4>(sA, 128, W1, b1, sB, 128, 128, 128, 64);
    __syncthreads();
    const int hi = lane >> 4, nl = lane & 15;
    for (int n0 = 0; n0 < 64; n0 += 16) {
        v8f acc[4];
        gemm_acc_mt<4>(sB, 128, W2, 64, b2[n0 + nl], 128, n0, acc);
#pragma unroll
        for (int t = 0; t < 4; ++t)
#pragma unroll
            for (int r = 0; r < 8; ++r) {
                int row = base + t * 16 + r + hi * 8;
                if (row < NNODES) {
                    int   n  = n0 + nl;
                    float dh = acc[t][r];
                    float ho = (float)h16[(size_t)row * 64 + n];
                    h16[(size_t)row * 64 + n] =
                        (half_t)(0.5f * ho + 0.5f * fmaxf(dh, 0.0f));
                }
            }
    }
}

// ---------------------------------------------------------------------------
// Final edge-weight MLP: w_in = [h[src], h[dst], e1, e2, e3] -> 128 -> 128 -> 1.
// 64 edges per wave. sB aliases sIn.
// ---------------------------------------------------------------------------
__global__ __launch_bounds__(32) void k_final(
    const int* __restrict__ ei, const half_t* __restrict__ h16,
    const half_t* __restrict__ e1, const half_t* __restrict__ e2,
    const half_t* __restrict__ e3,
    const half_t* __restrict__ W0, const float* __restrict__ b0,
    const half_t* __restrict__ W1, const float* __restrict__ b1,
    const float* __restrict__ W2, const float* __restrict__ b2,
    float* __restrict__ wout)
{
    __shared__ int sSrc[64], sDst[64];
    __shared__ __align__(16) half_t sIn[64 * 320];   // also reused as sB
    __shared__ __align__(16) half_t sA[64 * 128];
    half_t* sB = sIn;
    const int lane = threadIdx.x;
    const int base = blockIdx.x * 64;
    for (int i = lane; i < 64; i += 32) {
        sSrc[i] = ei[base + i];
        sDst[i] = ei[NEDGES + base + i];
    }
    __syncthreads();
    // gather concat input: 40 16-byte chunks per row
    for (int idx = lane; idx < 64 * 40; idx += 32) {
        int m = idx / 40, c = idx - m * 40;
        uint4 v;
        if (c < 8)       v = ((const uint4*)(h16 + (size_t)sSrc[m] * 64))[c];
        else if (c < 16) v = ((const uint4*)(h16 + (size_t)sDst[m] * 64))[c - 8];
        else if (c < 24) v = ((const uint4*)(e1 + (size_t)(base + m) * 64))[c - 16];
        else if (c < 32) v = ((const uint4*)(e2 + (size_t)(base + m) * 64))[c - 24];
        else             v = ((const uint4*)(e3 + (size_t)(base + m) * 64))[c - 32];
        ((uint4*)(sIn + m * 320))[c] = v;
    }
    __syncthreads();
    gemm_relu_mt<4>(sIn, 320, W0, b0, sA, 128, 320, 128, 64);
    __syncthreads();
    gemm_relu_mt<4>(sA, 128, W1, b1, sB, 128, 128, 128, 64);
    __syncthreads();
    // last layer N=1: plain dot per row, 2 rows per lane
#pragma unroll
    for (int rr = 0; rr < 2; ++rr) {
        int row = rr * 32 + lane;
        float s = b2[0];
        for (int k = 0; k < 128; ++k) s += (float)sB[row * 128 + k] * W2[k];
        wout[base + row] = 1.0f / (1.0f + __expf(-s));
    }
}

// --------------------------- utility kernels -------------------------------
__global__ void k_f2h(const float* __restrict__ s, half_t* __restrict__ d, int n) {
    int i = blockIdx.x * blockDim.x + threadIdx.x;
    if (i < n) d[i] = (half_t)s[i];
}
__global__ void k_h2f(const half_t* __restrict__ s, float* __restrict__ d, long long n) {
    long long i = (long long)blockIdx.x * blockDim.x + threadIdx.x;
    if (i < n) d[i] = (float)s[i];
}
__global__ void k_zero(float* __restrict__ p, long long n) {
    long long i = (long long)blockIdx.x * blockDim.x + threadIdx.x;
    if (i < n) p[i] = 0.0f;
}

// ---------------------------------------------------------------------------
extern "C" void kernel_launch(void* const* d_in, const int* in_sizes, int n_in,
                              void* d_out, int out_size, void* d_ws, size_t ws_size,
                              hipStream_t stream)
{
    (void)in_sizes; (void)n_in; (void)out_size; (void)ws_size;

    const float* x  = (const float*)d_in[0];
    const int*   ei = (const int*)d_in[1];
    const float* ea = (const float*)d_in[2];
    const float* neW0 = (const float*)d_in[3];
    const float* neW1 = (const float*)d_in[4];
    const float* eeW0 = (const float*)d_in[5];
    const float* eeW1 = (const float*)d_in[6];
    const float *rW0[3], *rB0[3], *rW1[3], *rB1[3], *rW2[3], *rB2[3];
    const float *oW0[3], *oB0[3], *oW1[3], *oB1[3], *oW2[3], *oB2[3];
    for (int l = 0; l < 3; ++l) {
        int B = 7 + l * 12;
        rW0[l] = (const float*)d_in[B + 0];  rB0[l] = (const float*)d_in[B + 1];
        rW1[l] = (const float*)d_in[B + 2];  rB1[l] = (const float*)d_in[B + 3];
        rW2[l] = (const float*)d_in[B + 4];  rB2[l] = (const float*)d_in[B + 5];
        oW0[l] = (const float*)d_in[B + 6];  oB0[l] = (const float*)d_in[B + 7];
        oW1[l] = (const float*)d_in[B + 8];  oB1[l] = (const float*)d_in[B + 9];
        oW2[l] = (const float*)d_in[B + 10]; oB2[l] = (const float*)d_in[B + 11];
    }
    const float* fW0 = (const float*)d_in[43]; const float* fB0 = (const float*)d_in[44];
    const float* fW1 = (const float*)d_in[45]; const float* fB1 = (const float*)d_in[46];
    const float* fW2 = (const float*)d_in[47]; const float* fB2 = (const float*)d_in[48];

    // ---- workspace carve ----
    char* wsb = (char*)d_ws;
    size_t off = 0;
    auto carve = [&](size_t bytes) -> char* {
        char* p = wsb + off;
        off += (bytes + 255) & ~(size_t)255;
        return p;
    };
    auto carve_h = [&](size_t elems) -> half_t* { return (half_t*)carve(elems * sizeof(half_t)); };

    half_t* hneW0 = carve_h(14 * 128);
    half_t* hneW1 = carve_h(128 * 64);
    half_t* heeW0 = carve_h(4 * 128);
    half_t* heeW1 = carve_h(128 * 64);
    half_t *hrW0[3], *hrW1[3], *hrW2[3], *hoW0[3], *hoW1[3], *hoW2[3];
    for (int l = 0; l < 3; ++l) {
        hrW0[l] = carve_h(192 * 128); hrW1[l] = carve_h(128 * 128); hrW2[l] = carve_h(128 * 64);
        hoW0[l] = carve_h(128 * 128); hoW1[l] = carve_h(128 * 128); hoW2[l] = carve_h(128 * 64);
    }
    half_t* hfW0 = carve_h(320 * 128);
    half_t* hfW1 = carve_h(128 * 128);

    half_t* h16 = carve_h((size_t)NNODES * 64);
    half_t* e16[4];
    for (int i = 0; i < 4; ++i) e16[i] = carve_h((size_t)NEDGES * 64);
    float* agg = (float*)carve((size_t)NNODES * 64 * sizeof(float));

    // ---- convert weights fp32 -> fp16 ----
    auto conv = [&](const float* s, half_t* d, int n) {
        k_f2h<<<(n + 255) / 256, 256, 0, stream>>>(s, d, n);
    };
    conv(neW0, hneW0, 14 * 128);  conv(neW1, hneW1, 128 * 64);
    conv(eeW0, heeW0, 4 * 128);   conv(eeW1, heeW1, 128 * 64);
    for (int l = 0; l < 3; ++l) {
        conv(rW0[l], hrW0[l], 192 * 128); conv(rW1[l], hrW1[l], 128 * 128);
        conv(rW2[l], hrW2[l], 128 * 64);
        conv(oW0[l], hoW0[l], 128 * 128); conv(oW1[l], hoW1[l], 128 * 128);
        conv(oW2[l], hoW2[l], 128 * 64);
    }
    conv(fW0, hfW0, 320 * 128);   conv(fW1, hfW1, 128 * 128);

    const int nodeBlocks = (NNODES + 63) / 64;   // 782 (tail-guarded)
    const int edgeBlocks = NEDGES / 64;          // 12500 (exact)

    // ---- encoders ----
    k_encoder<<<nodeBlocks, 32, 0, stream>>>(x, NODE_IN, NNODES, hneW0, hneW1, h16);
    k_encoder<<<edgeBlocks, 32, 0, stream>>>(ea, EDGE_IN, NEDGES, heeW0, heeW1, e16[0]);

    // ---- 3 interaction layers ----
    const long long aggN = (long long)NNODES * 64;
    for (int l = 0; l < 3; ++l) {
        k_zero<<<(unsigned)((aggN + 255) / 256), 256, 0, stream>>>(agg, aggN);
        k_edge_layer<<<edgeBlocks, 32, 0, stream>>>(
            ei, h16, e16[l],
            hrW0[l], rB0[l], hrW1[l], rB1[l], hrW2[l], rB2[l],
            agg, e16[l + 1]);
        k_node_layer<<<nodeBlocks, 32, 0, stream>>>(
            h16, agg,
            hoW0[l], oB0[l], hoW1[l], oB1[l], hoW2[l], oB2[l]);
    }

    // ---- final edge weights + outputs (w | h | e) ----
    float* out_w = (float*)d_out;
    float* out_h = out_w + NEDGES;
    float* out_e = out_h + (size_t)NNODES * 64;
    k_final<<<edgeBlocks, 32, 0, stream>>>(
        ei, h16, e16[1], e16[2], e16[3],
        hfW0, fB0, hfW1, fB1, fW2, fB2, out_w);

    const long long hN = (long long)NNODES * 64;
    const long long eN = (long long)NEDGES * 64;
    k_h2f<<<(unsigned)((hN + 255) / 256), 256, 0, stream>>>(h16, out_h, hN);
    k_h2f<<<(unsigned)((eN + 255) / 256), 256, 0, stream>>>(e16[3], out_e, eN);
}
